// MultiHeadedAttention_36825049596272
// MI455X (gfx1250) — compile-verified, matrix operand order checked by reading, not run on hardware
//
#include <hip/hip_runtime.h>
#include <hip/hip_bf16.h>

typedef __bf16 bf16_t;
typedef __attribute__((ext_vector_type(16))) __bf16 v16bf;
typedef __attribute__((ext_vector_type(8)))  float  v8f;
typedef __attribute__((ext_vector_type(4)))  int    v4i;

#define DEV __device__ __forceinline__

DEV int lane_id() { return (int)(threadIdx.x & 31); }

// ---- WMMA fragment loaders (CDNA5 16-bit layouts, CDNA5 ISA) ----
// A (16x32 MxK) / B-from-transposed-source:
//   lane 0-15: M=lane, K = {0..7, 16..23};  lane 16-31: M=lane-16, K = {8..15, 24..31}

DEV v16bf frag_rowmajor_f32(const float* __restrict__ p, int ld, int row0, int k0) {
  int lane = lane_id();
  const float* r = p + (size_t)(row0 + (lane & 15)) * ld + k0 + ((lane & 16) ? 8 : 0);
  v16bf f;
#pragma unroll
  for (int i = 0; i < 8; ++i) f[i] = (__bf16)r[i];
#pragma unroll
  for (int i = 0; i < 8; ++i) f[i + 8] = (__bf16)r[i + 16];
  return f;
}

DEV v16bf frag_rowmajor_bf16(const bf16_t* __restrict__ p, int ld, int row0, int k0) {
  int lane = lane_id();
  const bf16_t* r = p + (size_t)(row0 + (lane & 15)) * ld + k0 + ((lane & 16) ? 8 : 0);
  v16bf f;
#pragma unroll
  for (int i = 0; i < 8; ++i) f[i] = r[i];
#pragma unroll
  for (int i = 0; i < 8; ++i) f[i + 8] = r[i + 16];
  return f;
}

// B fragment from a K-row-major matrix (V): B[k][n] = p[(k0+k)*ld + n0+n].
// Only K=0..15 valid; upper half zero (pairs with zero-padded A).
DEV v16bf frag_b_k16_bf16(const bf16_t* __restrict__ p, int ld, int k0, int n0) {
  int lane = lane_id();
  int hi = (lane & 16) ? 8 : 0;
  int n = n0 + (lane & 15);
  v16bf f;
#pragma unroll
  for (int i = 0; i < 8; ++i) f[i] = p[(size_t)(k0 + hi + i) * ld + n];
#pragma unroll
  for (int i = 0; i < 8; ++i) f[i + 8] = (__bf16)0.0f;
  return f;
}

// fragment from an LDS tile, row stride `ld` bf16
DEV v16bf frag_lds(const bf16_t* base, int ld, int row0, int k0) {
  int lane = lane_id();
  const bf16_t* r = base + (size_t)(row0 + (lane & 15)) * ld + k0 + ((lane & 16) ? 8 : 0);
  v16bf f;
#pragma unroll
  for (int i = 0; i < 8; ++i) f[i] = r[i];
#pragma unroll
  for (int i = 0; i < 8; ++i) f[i + 8] = r[i + 16];
  return f;
}

DEV v8f wmma_bf16(v16bf a, v16bf b, v8f c) {
  return __builtin_amdgcn_wmma_f32_16x16x32_bf16(false, a, false, b, (short)0, c,
                                                 false, false);
}

// 16-byte global->LDS stage: gfx1250 async path (ASYNCcnt) when available.
DEV void stage16(const bf16_t* gsrc, bf16_t* ldst) {
#if __has_builtin(__builtin_amdgcn_global_load_async_to_lds_b128)
  __builtin_amdgcn_global_load_async_to_lds_b128(
      (__attribute__((address_space(1))) v4i*)gsrc,
      (__attribute__((address_space(3))) v4i*)ldst, 0, 0);
#else
  *(uint4*)ldst = *(const uint4*)gsrc;
#endif
}

DEV void async_wait0() {
#if __has_builtin(__builtin_amdgcn_global_load_async_to_lds_b128)
#if __has_builtin(__builtin_amdgcn_s_wait_asynccnt)
  __builtin_amdgcn_s_wait_asynccnt(0);
#else
  asm volatile("s_wait_asynccnt 0" ::: "memory");
#endif
#endif
}

// ---------- Kernel 0: streaming f32 -> bf16 conversion (one pass) -----------
__global__ void __launch_bounds__(256)
cvt_f32_bf16(const float* __restrict__ x, bf16_t* __restrict__ y, int n) {
  int i = (int)(blockIdx.x * blockDim.x + threadIdx.x) * 8;
  if (i >= n) return;
  float4 a = *(const float4*)(x + i);
  float4 b = *(const float4*)(x + i + 4);
  bf16_t o[8] = {(bf16_t)a.x, (bf16_t)a.y, (bf16_t)a.z, (bf16_t)a.w,
                 (bf16_t)b.x, (bf16_t)b.y, (bf16_t)b.z, (bf16_t)b.w};
  *(uint4*)(y + i) = *(const uint4*)o;
}

// ---------- Kernel 1: Y = A @ W^T (+ bias), bf16 operands -------------------
// 8 waves/block share a 64-wide weight supertile. Double-buffered 64x64 LDS
// tiles staged with async-to-LDS so the DMA of tile k+1 runs under the 8 WMMAs
// of tile k; one s_wait_asynccnt + barrier per 64-deep K step.
template <typename OutT>
__global__ void __launch_bounds__(256)
gemm_wtile(const bf16_t* __restrict__ A, const bf16_t* __restrict__ W,
           const float* __restrict__ bias, OutT* __restrict__ Y, int N, int D) {
  __shared__ bf16_t wtile[2][64 * 64];       // 2 x 8 KB
  int tilesN = D / 64;                       // 16
  int tn = (int)blockIdx.x % tilesN;
  int tm = ((int)blockIdx.x / tilesN) * 8 + (int)(threadIdx.x >> 5);
  int tid = (int)threadIdx.x;
  int srow = tid >> 3;                       // 0..31
  int scol = (tid & 7) * 8;                  // 0,8,...,56 (bf16 elems)
  const bf16_t* wbase = W + (size_t)tn * 64 * D;

  // prologue: stage tile 0
  stage16(wbase + (size_t)srow * D + scol,        &wtile[0][srow * 64 + scol]);
  stage16(wbase + (size_t)(srow + 32) * D + scol, &wtile[0][(srow + 32) * 64 + scol]);
  async_wait0();
  __syncthreads();

  v8f acc[4] = {};
  int nsteps = D / 64;                       // 16
  for (int s = 0; s < nsteps; ++s) {
    int k0 = s * 64;
    if (s + 1 < nsteps) {                    // kick off next tile (other buffer)
      bf16_t* nb = wtile[(s + 1) & 1];
      stage16(wbase + (size_t)srow * D + k0 + 64 + scol,        nb + srow * 64 + scol);
      stage16(wbase + (size_t)(srow + 32) * D + k0 + 64 + scol, nb + (srow + 32) * 64 + scol);
    }
    const bf16_t* cur = wtile[s & 1];
    v16bf a0 = frag_rowmajor_bf16(A, D, tm * 16, k0);
    v16bf a1 = frag_rowmajor_bf16(A, D, tm * 16, k0 + 32);
#pragma unroll
    for (int t = 0; t < 4; ++t) {
      v16bf b0 = frag_lds(cur, 64, t * 16, 0);   // B[k][n] = W[n][k]
      v16bf b1 = frag_lds(cur, 64, t * 16, 32);
      acc[t] = wmma_bf16(a1, b1, wmma_bf16(a0, b0, acc[t]));
    }
    async_wait0();                           // next tile landed
    __syncthreads();                         // everyone done reading cur
  }

  int lane = lane_id();
  int rbase = tm * 16 + ((lane & 16) ? 8 : 0);
#pragma unroll
  for (int t = 0; t < 4; ++t) {
    int n = tn * 64 + t * 16 + (lane & 15);
    float bv = bias ? bias[n] : 0.0f;
#pragma unroll
    for (int v = 0; v < 8; ++v)
      Y[(size_t)(rbase + v) * D + n] = (OutT)(acc[t][v] + bv);
  }
}

// ------------- Kernel 2: mask @ mask^T -> nonzero flags (u8) ----------------
__global__ void __launch_bounds__(256)
mask_mm_kernel(const float* __restrict__ mask, unsigned char* __restrict__ flags,
               int B, int S, int M) {
  int wave = blockIdx.x * (blockDim.x >> 5) + (threadIdx.x >> 5);
  int tilesS = S / 16;
  int perB = tilesS * tilesS;
  int b = wave / perB;
  if (b >= B) return;
  int rem = wave % perB;
  int tq = rem / tilesS, tk = rem % tilesS;
  const float* mb = mask + (size_t)b * S * M;
  v8f acc = {};
  for (int k0 = 0; k0 < M; k0 += 32) {
    v16bf a  = frag_rowmajor_f32(mb, M, tq * 16, k0);
    v16bf bb = frag_rowmajor_f32(mb, M, tk * 16, k0);
    acc = wmma_bf16(a, bb, acc);
  }
  int lane = lane_id();
  int n = tk * 16 + (lane & 15);
  int rbase = tq * 16 + ((lane & 16) ? 8 : 0);
#pragma unroll
  for (int v = 0; v < 8; ++v)
    flags[((size_t)b * S + rbase + v) * S + n] = (acc[v] != 0.0f) ? 1 : 0;
}

// ------------- Kernel 3: fused flash attention per (h,b,16 q-rows) ----------
__global__ void __launch_bounds__(256)
attention_kernel(const bf16_t* __restrict__ Qb, const bf16_t* __restrict__ Kb,
                 const bf16_t* __restrict__ Vb, const unsigned char* __restrict__ flags,
                 bf16_t* __restrict__ Xb, int B, int S, int D, int H) {
  __shared__ bf16_t plds[8 * 16 * 32];  // one 16x32 P tile per wave
  int wslot = threadIdx.x >> 5;
  bf16_t* pw = plds + wslot * (16 * 32);
  int lane = lane_id();

  // zero the K=16..31 pad columns once; stays zero for all iterations
  for (int idx = lane; idx < 16 * 16; idx += 32)
    pw[(idx >> 4) * 32 + 16 + (idx & 15)] = (bf16_t)0.0f;
  asm volatile("s_wait_dscnt 0" ::: "memory");

  int wave = blockIdx.x * (blockDim.x >> 5) + wslot;
  int tilesS = S / 16;
  int qt = wave % tilesS;
  int hb = wave / tilesS;
  int b = hb % B;
  int h = hb / B;
  if (h >= H) return;

  const int dh = 64;
  const bf16_t* Qp = Qb + (size_t)b * S * D + h * dh;  // [S, dh] stride D
  const bf16_t* Kp = Kb + (size_t)b * S * D + h * dh;
  const bf16_t* Vp = Vb + (size_t)b * S * D + h * dh;
  const unsigned char* fl = flags + ((size_t)b * S + (size_t)qt * 16) * S;

  v16bf aq0 = frag_rowmajor_bf16(Qp, D, qt * 16, 0);   // loop invariant
  v16bf aq1 = frag_rowmajor_bf16(Qp, D, qt * 16, 32);

  float rm[8], rl[8];
#pragma unroll
  for (int v = 0; v < 8; ++v) { rm[v] = -3.0e38f; rl[v] = 0.0f; }
  v8f o[4] = {};

  int hi8 = (lane & 16) ? 8 : 0;
  int ncol = lane & 15;

  for (int kb = 0; kb < tilesS; ++kb) {
    v8f s = {};
    v16bf bk = frag_rowmajor_bf16(Kp, D, kb * 16, 0);
    s = wmma_bf16(aq0, bk, s);
    bk = frag_rowmajor_bf16(Kp, D, kb * 16, 32);
    s = wmma_bf16(aq1, bk, s);

    float sv[8];
#pragma unroll
    for (int v = 0; v < 8; ++v) {
      unsigned char f = fl[(size_t)(v + hi8) * S + kb * 16 + ncol];
      sv[v] = f ? s[v] * 0.125f : -1.0e9f;   // 1/sqrt(64) scale + mask
    }

    // online softmax; row m spans 16 lanes of one half (C layout)
#pragma unroll
    for (int v = 0; v < 8; ++v) {
      float tmax = sv[v];
#pragma unroll
      for (int off = 1; off < 16; off <<= 1)
        tmax = fmaxf(tmax, __shfl_xor(tmax, off, 16));
      float nm = fmaxf(rm[v], tmax);
      float alpha = __expf(rm[v] - nm);
      float p = __expf(sv[v] - nm);
      float tsum = p;
#pragma unroll
      for (int off = 1; off < 16; off <<= 1)
        tsum += __shfl_xor(tsum, off, 16);
      rl[v] = rl[v] * alpha + tsum;
      rm[v] = nm;
      sv[v] = p;
#pragma unroll
      for (int t = 0; t < 4; ++t) o[t][v] = o[t][v] * alpha;
    }

    // P: C-layout -> LDS row-major -> A fragment (K padded to 32)
#pragma unroll
    for (int v = 0; v < 8; ++v)
      pw[(v + hi8) * 32 + ncol] = (bf16_t)sv[v];
    asm volatile("s_wait_dscnt 0" ::: "memory");  // DS in-order per wave
    v16bf pa = frag_lds(pw, 32, 0, 0);

#pragma unroll
    for (int t = 0; t < 4; ++t) {
      v16bf bv = frag_b_k16_bf16(Vp, D, kb * 16, t * 16);
      o[t] = wmma_bf16(pa, bv, o[t]);
    }
  }

  // normalize + scatter into torch's contiguous().view(B,-1,H*d) layout:
  // flat index h*S*d + s*d + dd  ->  row (h*S+s)>>4, col ((h*S+s)&15)*64+dd
#pragma unroll
  for (int v = 0; v < 8; ++v) {
    float invl = 1.0f / rl[v];
    int sq = qt * 16 + v + hi8;
    int tix = h * S + sq;
    int r = tix >> 4;
    int cbase = (tix & 15) * dh;
#pragma unroll
    for (int t = 0; t < 4; ++t) {
      int dd = t * 16 + ncol;
      Xb[((size_t)b * S + r) * (size_t)D + cbase + dd] = (bf16_t)(o[t][v] * invl);
    }
  }
}

// ---------------------------------------------------------------------------
extern "C" void kernel_launch(void* const* d_in, const int* in_sizes, int n_in,
                              void* d_out, int out_size, void* d_ws, size_t ws_size,
                              hipStream_t stream) {
  (void)in_sizes; (void)n_in; (void)out_size; (void)ws_size;
  const float* query = (const float*)d_in[0];
  const float* key   = (const float*)d_in[1];
  const float* value = (const float*)d_in[2];
  const float* mask  = (const float*)d_in[3];
  const float* Wq = (const float*)d_in[4];  const float* bq = (const float*)d_in[5];
  const float* Wk = (const float*)d_in[6];  const float* bk = (const float*)d_in[7];
  const float* Wv = (const float*)d_in[8];  const float* bv = (const float*)d_in[9];
  const float* Wo = (const float*)d_in[10];
  float* out = (float*)d_out;

  const int B = 4, S = 1024, Dm = 1024, H = 16, M = 64;
  const size_t nBSD = (size_t)B * S * Dm;   // 4M elements
  const size_t nDD  = (size_t)Dm * Dm;      // 1M elements

  // Workspace (bf16 unless noted):
  // Xq,Xk,Xv (8MB each) | Wqb,Wkb,Wvb,Wob (2MB each) | Q,K,V,X (8MB each) | flags u8 (4MB)
  bf16_t* Xq  = (bf16_t*)d_ws;
  bf16_t* Xk  = Xq + nBSD;
  bf16_t* Xv  = Xk + nBSD;
  bf16_t* Wqb = Xv + nBSD;
  bf16_t* Wkb = Wqb + nDD;
  bf16_t* Wvb = Wkb + nDD;
  bf16_t* Wob = Wvb + nDD;
  bf16_t* Qb  = Wob + nDD;
  bf16_t* Kb  = Qb + nBSD;
  bf16_t* Vb  = Kb + nBSD;
  bf16_t* Xb  = Vb + nBSD;
  unsigned char* flags = (unsigned char*)(Xb + nBSD);

  dim3 blk(256);
  const int wpb = 8;

  // one-pass f32 -> bf16 conversions (inputs read 16x, weights 32x downstream)
  dim3 cg_in((int)(nBSD / 8 / 256));
  dim3 cg_w((int)(nDD / 8 / 256));
  cvt_f32_bf16<<<cg_in, blk, 0, stream>>>(query, Xq, (int)nBSD);
  cvt_f32_bf16<<<cg_in, blk, 0, stream>>>(key,   Xk, (int)nBSD);
  cvt_f32_bf16<<<cg_in, blk, 0, stream>>>(value, Xv, (int)nBSD);
  cvt_f32_bf16<<<cg_w,  blk, 0, stream>>>(Wq, Wqb, (int)nDD);
  cvt_f32_bf16<<<cg_w,  blk, 0, stream>>>(Wk, Wkb, (int)nDD);
  cvt_f32_bf16<<<cg_w,  blk, 0, stream>>>(Wv, Wvb, (int)nDD);
  cvt_f32_bf16<<<cg_w,  blk, 0, stream>>>(Wo, Wob, (int)nDD);

  // projections: grid = (tilesM/8) * tilesN = 32 * 16 = 512 blocks
  int tilesM = (B * S) / 16;
  dim3 ggrid((tilesM / wpb) * (Dm / 64));
  gemm_wtile<bf16_t><<<ggrid, blk, 0, stream>>>(Xq, Wqb, bq, Qb, B * S, Dm);
  gemm_wtile<bf16_t><<<ggrid, blk, 0, stream>>>(Xk, Wkb, bk, Kb, B * S, Dm);
  gemm_wtile<bf16_t><<<ggrid, blk, 0, stream>>>(Xv, Wvb, bv, Vb, B * S, Dm);

  int mWaves = B * (S / 16) * (S / 16);
  mask_mm_kernel<<<dim3(mWaves / wpb), blk, 0, stream>>>(mask, flags, B, S, M);

  int aWaves = H * B * (S / 16);
  attention_kernel<<<dim3(aWaves / wpb), blk, 0, stream>>>(Qb, Kb, Vb, flags, Xb,
                                                           B, S, Dm, H);

  gemm_wtile<float><<<ggrid, blk, 0, stream>>>(Xb, Wob, (const float*)nullptr, out,
                                               B * S, Dm);
}
